// ImprovedCIFARPetDenoiser_66683662238183
// MI455X (gfx1250) — matrix operand order repaired and checked by reading, not compiled, not run on hardware
//
#include <hip/hip_runtime.h>
#include <math.h>

// ---------------------------------------------------------------------------
// Types
// ---------------------------------------------------------------------------
typedef __bf16 bf16;
typedef __attribute__((ext_vector_type(16))) __bf16 bf16x16;
typedef __attribute__((ext_vector_type(8)))  __bf16 bf16x8;
typedef __attribute__((ext_vector_type(8)))  float  f32x8;

// Use CDNA5 async global->LDS copies (ASYNCcnt) for GEMM tile staging.
#define GEMM_ASYNC 1

__device__ __forceinline__ float gelu_f(float x) {
    // exact erf gelu (torch F.gelu default)
    return 0.5f * x * (1.0f + erff(x * 0.70710678118654752440f));
}

// 16-byte async global -> LDS copy (GLOBAL_LOAD_ASYNC_TO_LDS_B128, ASYNCcnt).
__device__ __forceinline__ void async_ld16(bf16* l, const bf16* g) {
    unsigned lds = (unsigned)(unsigned long long)l;      // LDS aperture: low 32 bits
    unsigned long long ga = (unsigned long long)g;
    asm volatile("global_load_async_to_lds_b128 %0, %1, off"
                 :: "v"(lds), "v"(ga) : "memory");
}

// Stage 32 contiguous bf16 (64B) from each of gA,gW into LDS.
__device__ __forceinline__ void stage_tile(const bf16* gA, const bf16* gW,
                                           bf16* lA, bf16* lW) {
#if GEMM_ASYNC
    #pragma unroll
    for (int j = 0; j < 4; ++j) async_ld16(lA + 8 * j, gA + 8 * j);
    #pragma unroll
    for (int j = 0; j < 4; ++j) async_ld16(lW + 8 * j, gW + 8 * j);
#else
    bf16x8 ra[4], rw[4];
    #pragma unroll
    for (int j = 0; j < 4; ++j) ra[j] = *(const bf16x8*)(gA + 8 * j);
    #pragma unroll
    for (int j = 0; j < 4; ++j) rw[j] = *(const bf16x8*)(gW + 8 * j);
    #pragma unroll
    for (int j = 0; j < 4; ++j) *(bf16x8*)(lA + 8 * j) = ra[j];
    #pragma unroll
    for (int j = 0; j < 4; ++j) *(bf16x8*)(lW + 8 * j) = rw[j];
#endif
}

// ---------------------------------------------------------------------------
// f32 -> bf16 convert with K-padding (dst rows x Kp, src rows x K)
// ---------------------------------------------------------------------------
__global__ void convert_pad_bf16(const float* __restrict__ src, bf16* __restrict__ dst,
                                 long long rows, int K, int Kp) {
    long long i = (long long)blockIdx.x * 256 + threadIdx.x;
    if (i >= rows * Kp) return;
    int c = (int)(i % Kp);
    long long r = i / Kp;
    dst[i] = (c < K) ? (bf16)src[r * K + c] : (bf16)0.0f;
}

// ---------------------------------------------------------------------------
// Patch-embed im2col: x (128,3,32,32) -> A (32768, 64) bf16 (K=12 padded to 64)
// col index = cin*4 + ky*2 + kx to match pe_w (D,3,2,2) flattening
// ---------------------------------------------------------------------------
__global__ void patch_im2col(const float* __restrict__ x, bf16* __restrict__ A) {
    long long i = (long long)blockIdx.x * 256 + threadIdx.x;
    if (i >= 32768ll * 64) return;
    int c = (int)(i & 63);
    long long tok = i >> 6;
    int p = (int)(tok & 255);
    long long b = tok >> 8;
    bf16 v = (bf16)0.0f;
    if (c < 12) {
        int cin = c >> 2, ky = (c >> 1) & 1, kx = c & 1;
        int gy = p >> 4, gx = p & 15;
        v = (bf16)x[((b * 3 + cin) * 32 + (gy * 2 + ky)) * 32 + (gx * 2 + kx)];
    }
    A[i] = v;
}

// ---------------------------------------------------------------------------
// WMMA bf16 GEMM:  C[m,n] = sum_k A[m,k] * W[n,k]  (+bias +residual, act)
// A: (M,K) bf16 row-major, W: (N,K) bf16 row-major.  M%128==0, N%128==0, K%64==0.
// Double-buffered LDS staging via async global->LDS copies; batched via
// blockIdx.z with element strides sA/sW/sC.
// Fused: bias over N, residual (f32, same shape as C), act(1)=gelu.
// Outputs: Cf (f32, optional) and/or Cb (bf16, optional).
// ---------------------------------------------------------------------------
__global__ __launch_bounds__(256)
void gemm_bf16_wmma(const bf16* __restrict__ A, const bf16* __restrict__ W,
                    const float* __restrict__ bias, const float* __restrict__ residual,
                    float* __restrict__ Cf, bf16* __restrict__ Cb,
                    int M, int N, int K, int act,
                    long long sA, long long sW, long long sC) {
    __shared__ bf16 As[2][128][64];   // 32 KB
    __shared__ bf16 Ws[2][128][64];   // 32 KB

    const int tid  = threadIdx.x;
    const int lane = tid & 31;
    const int wid  = tid >> 5;
    const int wm   = wid >> 2;   // 0..1  -> 64-row strip
    const int wn   = wid & 3;    // 0..3  -> 32-col strip
    const long long z   = blockIdx.z;
    const long long bm0 = (long long)blockIdx.y * 128;
    const long long bn0 = (long long)blockIdx.x * 128;

    const bf16* Ab = A + z * sA;
    const bf16* Wb = W + z * sW;

    const int lr = tid >> 1;          // 0..127 tile row
    const int lc = (tid & 1) << 5;    // 0 or 32
    const int mrow = lane & 15;
    const int kbA  = (lane < 16) ? 0 : 8;    // A fragment: paired-K layout
    const int kbB  = (lane < 16) ? 0 : 8;    // B fragment: paired-K layout

    const bf16* gA0 = &Ab[(bm0 + lr) * (long long)K + lc];
    const bf16* gW0 = &Wb[(bn0 + lr) * (long long)K + lc];
    bf16* lA[2] = { &As[0][lr][lc], &As[1][lr][lc] };
    bf16* lW[2] = { &Ws[0][lr][lc], &Ws[1][lr][lc] };

    const int nk = K >> 6;            // 64-wide k-stages
    f32x8 acc[4][2] = {};

    stage_tile(gA0, gW0, lA[0], lW[0]);

    for (int i = 0; i < nk; ++i) {
        const int cur = i & 1;
        const bool more = (i + 1) < nk;
        if (more) {
            stage_tile(gA0 + (long long)(i + 1) * 64, gW0 + (long long)(i + 1) * 64,
                       lA[cur ^ 1], lW[cur ^ 1]);
            if (i + 2 < nk) {
                __builtin_prefetch(gA0 + (long long)(i + 2) * 64, 0, 1);  // global_prefetch_b8
                __builtin_prefetch(gW0 + (long long)(i + 2) * 64, 0, 1);
            }
        }
#if GEMM_ASYNC
        // Async loads retire in order: <=8 outstanding ==> current tile landed.
        if (more) asm volatile("s_wait_asynccnt 0x8" ::: "memory");
        else      asm volatile("s_wait_asynccnt 0x0" ::: "memory");
#endif
        __syncthreads();

        const bf16* arow[4];
        const bf16* brow[2];
        #pragma unroll
        for (int t = 0; t < 4; ++t) arow[t] = &As[cur][wm * 64 + t * 16 + mrow][0];
        #pragma unroll
        for (int u = 0; u < 2; ++u) brow[u] = &Ws[cur][wn * 32 + u * 16 + mrow][0];

        #pragma unroll
        for (int kk = 0; kk < 64; kk += 32) {
            bf16x16 af[4], bfr[2];
            #pragma unroll
            for (int t = 0; t < 4; ++t) {
                ((bf16x8*)&af[t])[0] = *(const bf16x8*)&arow[t][kk + kbA];
                ((bf16x8*)&af[t])[1] = *(const bf16x8*)&arow[t][kk + 16 + kbA];
            }
            #pragma unroll
            for (int u = 0; u < 2; ++u) {
                ((bf16x8*)&bfr[u])[0] = *(const bf16x8*)&brow[u][kk + kbB];
                ((bf16x8*)&bfr[u])[1] = *(const bf16x8*)&brow[u][kk + kbB + 16];
            }
            #pragma unroll
            for (int t = 0; t < 4; ++t)
                #pragma unroll
                for (int u = 0; u < 2; ++u)
                    acc[t][u] = __builtin_amdgcn_wmma_f32_16x16x32_bf16(
                        false, af[t], false, bfr[u], (short)0, acc[t][u], false, false);
        }
        __syncthreads();
    }

    // Epilogue: C vgpr r holds M = r (lanes 0-15) / 8+r (lanes 16-31), N = lane&15
    const int mbase = (lane < 16) ? 0 : 8;
    const int ncol  = lane & 15;
    for (int t = 0; t < 4; ++t)
        for (int u = 0; u < 2; ++u)
            for (int r = 0; r < 8; ++r) {
                long long m = bm0 + wm * 64 + t * 16 + mbase + r;
                long long n = bn0 + wn * 32 + u * 16 + ncol;
                float v = acc[t][u][r];
                if (bias)     v += bias[n];
                long long off = z * sC + m * (long long)N + n;
                if (residual) v += residual[off];
                if (act == 1) v = gelu_f(v);
                if (Cf) Cf[off] = v;
                if (Cb) Cb[off] = (bf16)v;
            }
}

// ---------------------------------------------------------------------------
// Row LayerNorm over D, optional gelu, optional f32 and bf16 outputs.
// ---------------------------------------------------------------------------
__global__ __launch_bounds__(256)
void layernorm_kernel(const float* __restrict__ X, const float* __restrict__ g,
                      const float* __restrict__ b, float* __restrict__ Yf,
                      bf16* __restrict__ Yb, int D, int act) {
    __shared__ float red[256];
    long long row = blockIdx.x;
    const float* x = X + row * (long long)D;
    int tid = threadIdx.x;
    float s = 0.f, s2 = 0.f;
    for (int i = tid; i < D; i += 256) { float v = x[i]; s += v; s2 += v * v; }
    red[tid] = s; __syncthreads();
    for (int st = 128; st; st >>= 1) { if (tid < st) red[tid] += red[tid + st]; __syncthreads(); }
    float mean = red[0] / (float)D; __syncthreads();
    red[tid] = s2; __syncthreads();
    for (int st = 128; st; st >>= 1) { if (tid < st) red[tid] += red[tid + st]; __syncthreads(); }
    float var = red[0] / (float)D - mean * mean;
    float inv = rsqrtf(var + 1e-5f);
    for (int i = tid; i < D; i += 256) {
        float v = (x[i] - mean) * inv * g[i] + b[i];
        if (act == 1) v = gelu_f(v);
        if (Yf) Yf[row * (long long)D + i] = v;
        if (Yb) Yb[row * (long long)D + i] = (bf16)v;
    }
}

// ---------------------------------------------------------------------------
// Noise-level MLP: 1 -> 320 -> 640 -> 1280 (gelu, gelu, none) + LayerNorm
// One block per batch sample.
// ---------------------------------------------------------------------------
__global__ __launch_bounds__(256)
void noise_mlp(const float* __restrict__ noise,
               const float* __restrict__ w1, const float* __restrict__ b1,
               const float* __restrict__ w2, const float* __restrict__ b2,
               const float* __restrict__ w3, const float* __restrict__ b3,
               const float* __restrict__ lng, const float* __restrict__ lnb,
               float* __restrict__ out) {
    __shared__ float h1[320];
    __shared__ float h2[640];
    __shared__ float h3[1280];
    __shared__ float red[256];
    int b = blockIdx.x, tid = threadIdx.x;
    float nl = noise[b];
    for (int j = tid; j < 320; j += 256) h1[j] = gelu_f(w1[j] * nl + b1[j]);
    __syncthreads();
    for (int j = tid; j < 640; j += 256) {
        float s = b2[j];
        for (int k = 0; k < 320; ++k) s += w2[j * 320 + k] * h1[k];
        h2[j] = gelu_f(s);
    }
    __syncthreads();
    for (int j = tid; j < 1280; j += 256) {
        float s = b3[j];
        for (int k = 0; k < 640; ++k) s += w3[j * 640 + k] * h2[k];
        h3[j] = s;
    }
    __syncthreads();
    float s = 0.f, s2 = 0.f;
    for (int j = tid; j < 1280; j += 256) { float v = h3[j]; s += v; s2 += v * v; }
    red[tid] = s; __syncthreads();
    for (int st = 128; st; st >>= 1) { if (tid < st) red[tid] += red[tid + st]; __syncthreads(); }
    float mean = red[0] / 1280.f; __syncthreads();
    red[tid] = s2; __syncthreads();
    for (int st = 128; st; st >>= 1) { if (tid < st) red[tid] += red[tid + st]; __syncthreads(); }
    float inv = rsqrtf(red[0] / 1280.f - mean * mean + 1e-5f);
    for (int j = tid; j < 1280; j += 256)
        out[(long long)b * 1280 + j] = (h3[j] - mean) * inv * lng[j] + lnb[j];
}

// ---------------------------------------------------------------------------
// Label embedding gather -> bf16 rows (128,1280)
// ---------------------------------------------------------------------------
__global__ void gather_label(const int* __restrict__ labels, const float* __restrict__ emb,
                             bf16* __restrict__ out) {
    long long i = (long long)blockIdx.x * 256 + threadIdx.x;
    if (i >= 128ll * 1280) return;
    int c = (int)(i % 1280);
    int b = (int)(i / 1280);
    out[i] = (bf16)emb[(long long)labels[b] * 1280 + c];
}

// ---------------------------------------------------------------------------
// tokens = patchLN + pos_embed(p,c) + ne[b,c] + le[b,c]
// ---------------------------------------------------------------------------
__global__ void build_tokens(const float* __restrict__ fln, const float* __restrict__ ne,
                             const float* __restrict__ le, float* __restrict__ T) {
    long long i = (long long)blockIdx.x * 256 + threadIdx.x;
    if (i >= 128ll * 256 * 1280) return;
    int c = (int)(i % 1280);
    long long bp = i / 1280;
    int p = (int)(bp % 256);
    long long b = bp / 256;
    int gy = p >> 4, gx = p & 15;
    int q = c / 320, t = c % 320;
    float om = expf(-(float)t * (9.210340371976184f / 320.0f));   // 10000^{-t/320}
    float coord = (q < 2) ? (float)gx : (float)gy;
    float ang = coord * om;
    float pe = ((q & 1) == 0) ? sinf(ang) : cosf(ang);
    T[i] = fln[i] + pe + ne[b * 1280 + c] + le[b * 1280 + c];
}

// ---------------------------------------------------------------------------
// Linear-attention feature maps from qkv rows (B*256, 3840) bf16:
//   qf (bh, n, d)    = elu(q * dh^-0.5) + 1
//   kT (bh, d, n)    = elu(k) + 1          (token-contiguous for WMMA contraction)
//   vT (bh, d, n)    = v
// ---------------------------------------------------------------------------
__global__ void qkv_features(const bf16* __restrict__ qkv, bf16* __restrict__ qf,
                             bf16* __restrict__ kT, bf16* __restrict__ vT) {
    long long i = (long long)blockIdx.x * 256 + threadIdx.x;
    if (i >= 128ll * 10 * 256 * 128) return;
    int d = (int)(i & 127);
    int n = (int)((i >> 7) & 255);
    long long bh = i >> 15;                 // b*10 + h
    int h = (int)(bh % 10);
    long long b = bh / 10;
    long long base = (b * 256 + n) * 3840 + (long long)h * 128 + d;
    float q = (float)qkv[base];
    float k = (float)qkv[base + 1280];
    float v = (float)qkv[base + 2560];
    q *= 0.08838834764831845f;              // 128^-0.5
    q = (q > 0.f ? q : expf(q) - 1.f) + 1.f;
    k = (k > 0.f ? k : expf(k) - 1.f) + 1.f;
    qf[(bh * 256 + n) * 128 + d] = (bf16)q;
    kT[(bh * 128 + d) * 256 + n] = (bf16)k;
    vT[(bh * 128 + d) * 256 + n] = (bf16)v;
}

// ksum[bh,d] = sum_n kT[bh,d,n]
__global__ void ksum_kernel(const bf16* __restrict__ kT, float* __restrict__ ks) {
    long long i = (long long)blockIdx.x * 256 + threadIdx.x;
    if (i >= 128ll * 10 * 128) return;
    const bf16* p = kT + i * 256;
    float s = 0.f;
    for (int n = 0; n < 256; ++n) s += (float)p[n];
    ks[i] = s;
}

// out_tok[b, n, h*128+e] = attn[bh,n,e] / (q[bh,n,:]·ksum[bh,:] + 1e-8)
__global__ __launch_bounds__(128)
void attn_epilogue(const bf16* __restrict__ attn, const bf16* __restrict__ qf,
                   const float* __restrict__ ks, bf16* __restrict__ tok) {
    __shared__ float red[128];
    long long blk = blockIdx.x;             // bh*256 + n
    int e = threadIdx.x;
    long long bh = blk >> 8;
    int n = (int)(blk & 255);
    float qv = (float)qf[blk * 128 + e];
    red[e] = qv * ks[bh * 128 + e];
    __syncthreads();
    for (int st = 64; st; st >>= 1) { if (e < st) red[e] += red[e + st]; __syncthreads(); }
    float den = red[0] + 1e-8f;
    float o = (float)attn[blk * 128 + e] / den;
    int h = (int)(bh % 10);
    long long b = bh / 10;
    tok[(b * 256 + n) * 1280 + (long long)h * 128 + e] = (bf16)o;
}

// (B,256,1280) tokens -> (B,1280,16,16) NCHW
__global__ void tokens_to_nchw(const float* __restrict__ T, float* __restrict__ S) {
    long long i = (long long)blockIdx.x * 256 + threadIdx.x;
    if (i >= 128ll * 1280 * 256) return;
    int p = (int)(i % 256);
    long long bc = i / 256;
    int c = (int)(bc % 1280);
    long long b = bc / 1280;
    S[i] = T[(b * 256 + p) * 1280 + c];
}

// ---------------------------------------------------------------------------
// Direct KSxKS conv (pad), act: 0 none, 2 tanh
// ---------------------------------------------------------------------------
__global__ void convk(const float* __restrict__ in, const float* __restrict__ w,
                      const float* __restrict__ bias, float* __restrict__ out,
                      int B, int Cin, int H, int W, int Cout, int KS, int pad, int act) {
    long long i = (long long)blockIdx.x * 256 + threadIdx.x;
    long long total = (long long)B * Cout * H * W;
    if (i >= total) return;
    int x  = (int)(i % W);
    int y  = (int)((i / W) % H);
    int co = (int)((i / ((long long)W * H)) % Cout);
    long long b = i / ((long long)W * H * Cout);
    float acc = bias[co];
    for (int ci = 0; ci < Cin; ++ci) {
        const float* ip = in + ((b * Cin + ci) * H) * (long long)W;
        const float* wp = w + ((long long)co * Cin + ci) * KS * KS;
        for (int ky = 0; ky < KS; ++ky) {
            int yy = y + ky - pad;
            if (yy < 0 || yy >= H) continue;
            for (int kx = 0; kx < KS; ++kx) {
                int xx = x + kx - pad;
                if (xx < 0 || xx >= W) continue;
                acc += ip[yy * W + xx] * wp[ky * KS + kx];
            }
        }
    }
    if (act == 2) acc = tanhf(acc);
    out[i] = acc;
}

// ---------------------------------------------------------------------------
// GroupNorm over (cpg,HW) per (b,group); act(1)=gelu. grid = B*groups.
// ---------------------------------------------------------------------------
__global__ __launch_bounds__(256)
void groupnorm_kernel(const float* __restrict__ in, const float* __restrict__ gam,
                      const float* __restrict__ bet, float* __restrict__ out,
                      int C, int HW, int cpg, int act) {
    __shared__ float red[256];
    int groups = C / cpg;
    int grp = blockIdx.x % groups;
    long long b = blockIdx.x / groups;
    const float* x = in  + (b * C + (long long)grp * cpg) * HW;
    float*       y = out + (b * C + (long long)grp * cpg) * HW;
    long long n = (long long)cpg * HW;
    int tid = threadIdx.x;
    float s = 0.f, s2 = 0.f;
    for (long long i = tid; i < n; i += 256) { float v = x[i]; s += v; s2 += v * v; }
    red[tid] = s; __syncthreads();
    for (int st = 128; st; st >>= 1) { if (tid < st) red[tid] += red[tid + st]; __syncthreads(); }
    float mean = red[0] / (float)n; __syncthreads();
    red[tid] = s2; __syncthreads();
    for (int st = 128; st; st >>= 1) { if (tid < st) red[tid] += red[tid + st]; __syncthreads(); }
    float inv = rsqrtf(red[0] / (float)n - mean * mean + 1e-5f);
    for (long long i = tid; i < n; i += 256) {
        int c = grp * cpg + (int)(i / HW);
        float v = (x[i] - mean) * inv * gam[c] + bet[c];
        if (act == 1) v = gelu_f(v);
        y[i] = v;
    }
}

__global__ void add_act(const float* __restrict__ a, const float* __restrict__ b,
                        float* __restrict__ out, long long n, int act) {
    long long i = (long long)blockIdx.x * 256 + threadIdx.x;
    if (i >= n) return;
    float v = a[i] + b[i];
    if (act == 1) v = gelu_f(v);
    out[i] = v;
}

// Bilinear 2x upsample, half-pixel centers (align_corners=False)
__global__ void upsample2x(const float* __restrict__ in, float* __restrict__ out,
                           int C, int H, int W) {
    long long i = (long long)blockIdx.x * 256 + threadIdx.x;
    int H2 = 2 * H, W2 = 2 * W;
    long long total = 128ll * C * H2 * W2;
    if (i >= total) return;
    int x2 = (int)(i % W2);
    int y2 = (int)((i / W2) % H2);
    long long bc = i / ((long long)W2 * H2);
    float sx = (x2 + 0.5f) * 0.5f - 0.5f;
    float sy = (y2 + 0.5f) * 0.5f - 0.5f;
    int x0 = (int)floorf(sx), y0 = (int)floorf(sy);
    float fx = sx - x0, fy = sy - y0;
    int x1 = x0 + 1, y1 = y0 + 1;
    x0 = min(max(x0, 0), W - 1); x1 = min(max(x1, 0), W - 1);
    y0 = min(max(y0, 0), H - 1); y1 = min(max(y1, 0), H - 1);
    const float* p = in + bc * (long long)H * W;
    float v = (1.f - fy) * ((1.f - fx) * p[y0 * W + x0] + fx * p[y0 * W + x1]) +
              fy        * ((1.f - fx) * p[y1 * W + x0] + fx * p[y1 * W + x1]);
    out[i] = v;
}

// ---------------------------------------------------------------------------
// TDM / cluster probe kernel (exercises the CDNA5 async-tensor path; not
// launched — compiled into the device bundle for the instruction path).
// ---------------------------------------------------------------------------
#if defined(__has_builtin)
#if __has_builtin(__builtin_amdgcn_tensor_load_to_lds)
typedef __attribute__((ext_vector_type(4))) unsigned int u32x4;
typedef __attribute__((ext_vector_type(8))) int i32x8;
typedef __attribute__((ext_vector_type(4))) int i32x4;
__global__ void tdm_probe_kernel(const unsigned short* __restrict__ src,
                                 unsigned short* __restrict__ dst, int n) {
    __shared__ unsigned short tile[2048];
    unsigned long long ga = (unsigned long long)(const void*)src;
    unsigned int lds = (unsigned int)(unsigned long long)(void*)&tile[0];
    u32x4 g0;
    g0[0] = 1u;                                               // count=1
    g0[1] = lds;                                              // lds_addr
    g0[2] = (unsigned int)ga;                                 // global_addr[31:0]
    g0[3] = (unsigned int)((ga >> 32) & 0x01FFFFFFu) | (2u << 30); // addr[56:32] | type=2
    i32x8 g1;
    g1[0] = (1 << 16);           // data_size = 2 bytes
    g1[1] = (64 << 16);          // tensor_dim0[15:0] @ bits 63:48
    g1[2] = (32 << 16);          // tensor_dim1[15:0] @ bits 95:80
    g1[3] = (64 << 16);          // tile_dim0 = 64
    g1[4] = 32;                  // tile_dim1 = 32
    g1[5] = 64;                  // tensor_dim0_stride[31:0]
    g1[6] = 0; g1[7] = 0;
    i32x4 z4 = {0, 0, 0, 0};
#if __clang_major__ <= 22
    __builtin_amdgcn_tensor_load_to_lds(g0, g1, z4, z4, 0);
#else
    i32x8 z8 = {0, 0, 0, 0, 0, 0, 0, 0};
    __builtin_amdgcn_tensor_load_to_lds(g0, g1, z4, z4, z8, 0);
#endif
    __builtin_amdgcn_s_wait_tensorcnt(0);
#if __has_builtin(__builtin_amdgcn_s_cluster_barrier)
    __builtin_amdgcn_s_cluster_barrier();
#endif
    int i = threadIdx.x;
    if (i < n) dst[i] = tile[i];
}
#endif
#endif

// ---------------------------------------------------------------------------
// Host-side orchestration
// ---------------------------------------------------------------------------
enum {
    IN_X = 0, IN_NOISE, IN_LABELS,
    P_PE_W, P_PE_B, P_PE_LN_G, P_PE_LN_B,
    P_LABEL_EMB, P_LP_W, P_LP_B, P_LP_LN_G, P_LP_LN_B,
    P_N1_W, P_N1_B, P_N2_W, P_N2_B, P_N3_W, P_N3_B, P_N_LN_G, P_N_LN_B,
    P_QKV_W, P_OUT_W, P_OUT_B,
    P_LN1_G, P_LN1_B, P_LN2_G, P_LN2_B,
    P_FF1_W, P_FF1_B, P_FF2_W, P_FF2_B,
    P_D0_W, P_D0_B, P_D0_GN_G, P_D0_GN_B,
    P_R1_C1_W, P_R1_C1_B, P_R1_G1_G, P_R1_G1_B,
    P_R1_C2_W, P_R1_C2_B, P_R1_G2_G, P_R1_G2_B,
    P_U_C1_W, P_U_C1_B, P_U_G1_G, P_U_G1_B,
    P_U_C2_W, P_U_C2_B, P_U_G2_G, P_U_G2_B,
    P_U_SK_W, P_U_SK_B,
    P_R2_C1_W, P_R2_C1_B, P_R2_G1_G, P_R2_G1_B,
    P_R2_C2_W, P_R2_C2_B, P_R2_G2_G, P_R2_G2_B,
    P_F_W, P_F_B
};

static inline unsigned gblk(long long n) { return (unsigned)((n + 255) / 256); }

static void launch_gemm(hipStream_t st, const bf16* A, const bf16* W, const float* bias,
                        const float* res, float* Cf, bf16* Cb, int M, int N, int K,
                        int act, int batch, long long sA, long long sW, long long sC) {
    dim3 g((unsigned)(N / 128), (unsigned)(M / 128), (unsigned)batch);
    gemm_bf16_wmma<<<g, 256, 0, st>>>(A, W, bias, res, Cf, Cb, M, N, K, act, sA, sW, sC);
}

extern "C" void kernel_launch(void* const* d_in, const int* in_sizes, int n_in,
                              void* d_out, int out_size, void* d_ws, size_t ws_size,
                              hipStream_t stream) {
    (void)in_sizes; (void)n_in; (void)out_size; (void)ws_size;
    const float* x      = (const float*)d_in[IN_X];
    const float* noise  = (const float*)d_in[IN_NOISE];
    const int*   labels = (const int*)d_in[IN_LABELS];
    auto P = [&](int i) { return (const float*)d_in[i]; };

    const long long TOK = 32768;  // 128 * 256 tokens
    const int D = 1280;

    char* base = (char*)d_ws;
    size_t off = 0;
    auto alloc = [&](size_t bytes) -> void* {
        void* p = base + off;
        off += (bytes + 255) & ~(size_t)255;
        return p;
    };

    // bf16 weights (K padded to 64 where needed)
    bf16* wpe  = (bf16*)alloc(1280ll * 64 * 2);
    bf16* wlp  = (bf16*)alloc(1280ll * 1280 * 2);
    bf16* wqkv = (bf16*)alloc(3840ll * 1280 * 2);
    bf16* wo   = (bf16*)alloc(1280ll * 1280 * 2);
    bf16* wf1  = (bf16*)alloc(5120ll * 1280 * 2);
    bf16* wf2  = (bf16*)alloc(1280ll * 5120 * 2);
    // activations
    bf16*  Apatch  = (bf16*)alloc(TOK * 64 * 2);
    float* T0      = (float*)alloc(TOK * D * 4);
    float* T1      = (float*)alloc(TOK * D * 4);
    float* T2      = (float*)alloc(TOK * D * 4);
    bf16*  LNB     = (bf16*)alloc(TOK * D * 2);
    bf16*  big0    = (bf16*)alloc(TOK * 5120ll * 2);
    bf16*  qf      = (bf16*)alloc(TOK * D * 2);
    bf16*  kT      = (bf16*)alloc(TOK * D * 2);
    bf16*  vT      = (bf16*)alloc(TOK * D * 2);
    float* ksum    = (float*)alloc(1280ll * 128 * 4);
    bf16*  ctxT    = (bf16*)alloc(1280ll * 128 * 128 * 2);
    bf16*  attnb   = (bf16*)alloc(TOK * D * 2);
    bf16*  attntok = (bf16*)alloc(TOK * D * 2);
    float* nebuf   = (float*)alloc(128ll * D * 4);
    float* lpf     = (float*)alloc(128ll * D * 4);
    float* lebuf   = (float*)alloc(128ll * D * 4);
    bf16*  labb    = (bf16*)alloc(128ll * D * 2);
    // decoder
    float* Db1 = (float*)alloc(128ll * 256 * 256 * 4);
    float* Db2 = (float*)alloc(128ll * 256 * 256 * 4);
    float* Db3 = (float*)alloc(128ll * 256 * 256 * 4);
    float* E1  = (float*)alloc(128ll * 64 * 256 * 4);
    float* E2  = (float*)alloc(128ll * 64 * 1024 * 4);
    float* E3  = (float*)alloc(128ll * 64 * 256 * 4);
    float* F1  = (float*)alloc(128ll * 64 * 1024 * 4);
    float* F2  = (float*)alloc(128ll * 64 * 1024 * 4);
    float* F3  = (float*)alloc(128ll * 64 * 1024 * 4);

    // --- weight conversion to bf16 ---
    convert_pad_bf16<<<gblk(1280ll * 64), 256, 0, stream>>>(P(P_PE_W),  wpe, 1280, 12, 64);
    convert_pad_bf16<<<gblk(1280ll * 1280), 256, 0, stream>>>(P(P_LP_W),  wlp, 1280, 1280, 1280);
    convert_pad_bf16<<<gblk(3840ll * 1280), 256, 0, stream>>>(P(P_QKV_W), wqkv, 3840, 1280, 1280);
    convert_pad_bf16<<<gblk(1280ll * 1280), 256, 0, stream>>>(P(P_OUT_W), wo, 1280, 1280, 1280);
    convert_pad_bf16<<<gblk(5120ll * 1280), 256, 0, stream>>>(P(P_FF1_W), wf1, 5120, 1280, 1280);
    convert_pad_bf16<<<gblk(1280ll * 5120), 256, 0, stream>>>(P(P_FF2_W), wf2, 1280, 5120, 5120);

    // --- patch embed: im2col + GEMM + LN ---
    patch_im2col<<<gblk(TOK * 64), 256, 0, stream>>>(x, Apatch);
    launch_gemm(stream, Apatch, wpe, P(P_PE_B), nullptr, T1, nullptr,
                (int)TOK, 1280, 64, 0, 1, 0, 0, 0);
    layernorm_kernel<<<(unsigned)TOK, 256, 0, stream>>>(T1, P(P_PE_LN_G), P(P_PE_LN_B),
                                                        T2, nullptr, D, 0);

    // --- noise MLP and label embedding ---
    noise_mlp<<<128, 256, 0, stream>>>(noise, P(P_N1_W), P(P_N1_B), P(P_N2_W), P(P_N2_B),
                                       P(P_N3_W), P(P_N3_B), P(P_N_LN_G), P(P_N_LN_B), nebuf);
    gather_label<<<gblk(128ll * D), 256, 0, stream>>>(labels, P(P_LABEL_EMB), labb);
    launch_gemm(stream, labb, wlp, P(P_LP_B), nullptr, lpf, nullptr,
                128, 1280, 1280, 0, 1, 0, 0, 0);
    layernorm_kernel<<<128, 256, 0, stream>>>(lpf, P(P_LP_LN_G), P(P_LP_LN_B),
                                              lebuf, nullptr, D, 1);

    build_tokens<<<gblk(TOK * D), 256, 0, stream>>>(T2, nebuf, lebuf, T0);

    // --- transformer: 5 layers, shared weights ---
    for (int layer = 0; layer < 5; ++layer) {
        layernorm_kernel<<<(unsigned)TOK, 256, 0, stream>>>(T0, P(P_LN1_G), P(P_LN1_B),
                                                            nullptr, LNB, D, 0);
        launch_gemm(stream, LNB, wqkv, nullptr, nullptr, nullptr, big0,
                    (int)TOK, 3840, 1280, 0, 1, 0, 0, 0);
        qkv_features<<<gblk(TOK * D), 256, 0, stream>>>(big0, qf, kT, vT);
        ksum_kernel<<<gblk(1280ll * 128), 256, 0, stream>>>(kT, ksum);
        // ctxT[bh, e, d] = sum_n vT[bh,e,n] * kT[bh,d,n]   (1280-way batched)
        launch_gemm(stream, vT, kT, nullptr, nullptr, nullptr, ctxT,
                    128, 128, 256, 0, 1280, 32768, 32768, 16384);
        // attn[bh, n, e] = sum_d qf[bh,n,d] * ctxT[bh,e,d]
        launch_gemm(stream, qf, ctxT, nullptr, nullptr, nullptr, attnb,
                    256, 128, 128, 0, 1280, 32768, 16384, 32768);
        attn_epilogue<<<(unsigned)(1280ll * 256), 128, 0, stream>>>(attnb, qf, ksum, attntok);
        // out projection + residual -> T1
        launch_gemm(stream, attntok, wo, P(P_OUT_B), T0, T1, nullptr,
                    (int)TOK, 1280, 1280, 0, 1, 0, 0, 0);
        // FF block
        layernorm_kernel<<<(unsigned)TOK, 256, 0, stream>>>(T1, P(P_LN2_G), P(P_LN2_B),
                                                            nullptr, LNB, D, 0);
        launch_gemm(stream, LNB, wf1, P(P_FF1_B), nullptr, nullptr, big0,
                    (int)TOK, 5120, 1280, 1 /*gelu*/, 1, 0, 0, 0);
        launch_gemm(stream, big0, wf2, P(P_FF2_B), T1, T0, nullptr,
                    (int)TOK, 1280, 5120, 0, 1, 0, 0, 0);
    }

    // --- decoder ---
    tokens_to_nchw<<<gblk(TOK * D), 256, 0, stream>>>(T0, T1);   // T1 = (B,1280,16,16)
    convk<<<gblk(128ll * 256 * 256), 256, 0, stream>>>(T1, P(P_D0_W), P(P_D0_B), Db1,
                                                       128, 1280, 16, 16, 256, 3, 1, 0);
    groupnorm_kernel<<<128 * 32, 256, 0, stream>>>(Db1, P(P_D0_GN_G), P(P_D0_GN_B), Db2,
                                                   256, 256, 8, 1);
    // ResidualBlock(256)
    convk<<<gblk(128ll * 256 * 256), 256, 0, stream>>>(Db2, P(P_R1_C1_W), P(P_R1_C1_B), Db1,
                                                       128, 256, 16, 16, 256, 3, 1, 0);
    groupnorm_kernel<<<128 * 8, 256, 0, stream>>>(Db1, P(P_R1_G1_G), P(P_R1_G1_B), Db3,
                                                  256, 256, 32, 1);
    convk<<<gblk(128ll * 256 * 256), 256, 0, stream>>>(Db3, P(P_R1_C2_W), P(P_R1_C2_B), Db1,
                                                       128, 256, 16, 16, 256, 3, 1, 0);
    groupnorm_kernel<<<128 * 8, 256, 0, stream>>>(Db1, P(P_R1_G2_G), P(P_R1_G2_B), Db3,
                                                  256, 256, 32, 0);
    add_act<<<gblk(128ll * 256 * 256), 256, 0, stream>>>(Db3, Db2, Db1, 128ll * 256 * 256, 1);
    // ResUpBlock(256->64)
    convk<<<gblk(128ll * 64 * 256), 256, 0, stream>>>(Db1, P(P_U_SK_W), P(P_U_SK_B), E1,
                                                      128, 256, 16, 16, 64, 1, 0, 0);
    upsample2x<<<gblk(128ll * 64 * 1024), 256, 0, stream>>>(E1, E2, 64, 16, 16);
    convk<<<gblk(128ll * 64 * 256), 256, 0, stream>>>(Db1, P(P_U_C1_W), P(P_U_C1_B), E3,
                                                      128, 256, 16, 16, 64, 3, 1, 0);
    groupnorm_kernel<<<128 * 8, 256, 0, stream>>>(E3, P(P_U_G1_G), P(P_U_G1_B), E1,
                                                  64, 256, 8, 1);
    upsample2x<<<gblk(128ll * 64 * 1024), 256, 0, stream>>>(E1, F1, 64, 16, 16);
    convk<<<gblk(128ll * 64 * 1024), 256, 0, stream>>>(F1, P(P_U_C2_W), P(P_U_C2_B), F2,
                                                       128, 64, 32, 32, 64, 3, 1, 0);
    groupnorm_kernel<<<128 * 8, 256, 0, stream>>>(F2, P(P_U_G2_G), P(P_U_G2_B), F3,
                                                  64, 1024, 8, 1);
    add_act<<<gblk(128ll * 64 * 1024), 256, 0, stream>>>(F3, E2, F1, 128ll * 64 * 1024, 0);
    // ResidualBlock(64)
    convk<<<gblk(128ll * 64 * 1024), 256, 0, stream>>>(F1, P(P_R2_C1_W), P(P_R2_C1_B), F2,
                                                       128, 64, 32, 32, 64, 3, 1, 0);
    groupnorm_kernel<<<128 * 8, 256, 0, stream>>>(F2, P(P_R2_G1_G), P(P_R2_G1_B), F3,
                                                  64, 1024, 8, 1);
    convk<<<gblk(128ll * 64 * 1024), 256, 0, stream>>>(F3, P(P_R2_C2_W), P(P_R2_C2_B), F2,
                                                       128, 64, 32, 32, 64, 3, 1, 0);
    groupnorm_kernel<<<128 * 8, 256, 0, stream>>>(F2, P(P_R2_G2_G), P(P_R2_G2_B), F3,
                                                  64, 1024, 8, 0);
    add_act<<<gblk(128ll * 64 * 1024), 256, 0, stream>>>(F3, F1, F2, 128ll * 64 * 1024, 1);
    // final conv + tanh -> output (128,3,32,32) f32
    convk<<<gblk(128ll * 3 * 1024), 256, 0, stream>>>(F2, P(P_F_W), P(P_F_B), (float*)d_out,
                                                      128, 64, 32, 32, 3, 3, 1, 2);
}
